// BertAdapterCapsuleGrow_71459665871073
// MI455X (gfx1250) — compile-verified, matrix operand order checked by reading, not run on hardware
//
#include <hip/hip_runtime.h>
#include <cstdint>
#include <cstddef>

// ---------------- problem constants ----------------
#define BB    64
#define SS    128
#define HH    768
#define AA    512
#define CO    100
#define NN    64
#define SC    12
#define TT    5
#define T1    6
#define MROWS (BB * SS)     // 8192
#define BD    256

typedef __bf16 bf16_t;
typedef __attribute__((ext_vector_type(16))) __bf16 v16bf;
typedef __attribute__((ext_vector_type(8)))  float  v8f;

struct alignas(16) U4 { uint32_t a, b, c, d; };   // POD 16B vector (union-safe)

static __device__ __forceinline__ bf16_t f2bf(float f) {
  union { float f; uint32_t u; } c; c.f = f;
  uint32_t r = (c.u + 0x7FFFu + ((c.u >> 16) & 1u)) >> 16;   // RNE
  union { uint16_t s; bf16_t b; } o; o.s = (uint16_t)r;
  return o.b;
}

// ---------------- prep: f32 -> bf16 ----------------
__global__ void k_f32_to_bf16(const float* __restrict__ src, bf16_t* __restrict__ dst, int n) {
  int i = blockIdx.x * blockDim.x + threadIdx.x;
  if (i < n) dst[i] = f2bf(src[i]);
}

// conv weight: w[CO][H][K] -> wt[CO][K*H], wt[co][k*H+h] = w[co][h][k]  (bf16)
__global__ void k_convw(const float* __restrict__ w, bf16_t* __restrict__ wt, int Kw) {
  int i = blockIdx.x * blockDim.x + threadIdx.x;
  int tot = CO * HH * Kw;
  if (i >= tot) return;
  int co = i / (HH * Kw);
  int r  = i - co * HH * Kw;
  int h  = r / Kw;
  int k  = r - h * Kw;
  wt[(size_t)co * (Kw * HH) + (size_t)k * HH + h] = f2bf(w[i]);
}

// ---------------- generic bf16 WMMA GEMM ----------------
// C[b][m][n] = act( A[b][m][k] * B[b][n][k]^T + bias[n] (+ bias2[(b%mod)][n]) )
// Block tile 128(M) x 64(N), K step 32. 8 waves: 4(M) x 2(N), wave tile 32x32.
// Double-buffered LDS staging via GLOBAL_LOAD_ASYNC_TO_LDS_B128 (ASYNCcnt).
#define ASYNC_CP_B128(ldsaddr, gptr)                                        \
  asm volatile("global_load_async_to_lds_b128 %0, %1, off"                  \
               :: "v"(ldsaddr), "v"(gptr) : "memory")

__launch_bounds__(256)
__global__ void k_wmma_gemm(const bf16_t* __restrict__ A, long long sAb, int lda,
                            const bf16_t* __restrict__ Bm, long long sBb, int ldb,
                            float* __restrict__ Cf, bf16_t* __restrict__ Cb,
                            long long sCb, int ldc,
                            const float* __restrict__ bias, long long sBiasB,
                            const float* __restrict__ bias2, int bias2_mod, int bias2_ld,
                            int M, int N, int K, int act) {
  __shared__ bf16_t lA[2][128][40];   // +8 pad: 80B row stride, conflict-free frag loads
  __shared__ bf16_t lB[2][64][40];

  const int tid   = threadIdx.x;
  const int batch = blockIdx.z;
  const int mbase = blockIdx.y * 128;
  const int nbase = blockIdx.x * 64;

  const bf16_t* Ab = A + (size_t)batch * (size_t)sAb;
  const bf16_t* Bb = Bm + (size_t)batch * (size_t)sBb;

  // A tile loader: 256 thr x 16 bf16 = 128x32 (two b128 async copies each)
  const int ar = tid >> 1;
  const int ac = (tid & 1) << 4;
  int arow = mbase + ar; if (arow > M - 1) arow = M - 1;     // clamp (conv edge tiles)
  const bf16_t* aptr = Ab + (size_t)arow * (size_t)lda + ac;

  // B tile loader: 256 thr x 8 bf16 = 64x32 (one b128 async copy each)
  const int br = tid >> 2;
  const int bc = (tid & 3) << 3;
  int brow = nbase + br; if (brow > N - 1) brow = N - 1;     // clamp (N=100)
  const bf16_t* bptr = Bb + (size_t)brow * (size_t)ldb + bc;

  // LDS byte addresses for the async DMA (low 32 bits of flat addr = LDS offset)
  const uint32_t ldsA = (uint32_t)(uintptr_t)&lA[0][ar][ac];
  const uint32_t ldsB = (uint32_t)(uintptr_t)&lB[0][br][bc];
  const uint32_t strA = (uint32_t)sizeof(lA[0]);   // 10240 B
  const uint32_t strB = (uint32_t)sizeof(lB[0]);   // 5120 B

  const int lane = tid & 31;
  const int wave = tid >> 5;
  const int wm = (wave >> 1) << 5;   // 0,32,64,96
  const int wn = (wave & 1) << 5;    // 0,32
  const int lr = lane & 15;
  const int kh = (lane >> 4) << 3;   // K-half select per ISA A/B layout

  union FAcc { float f[8]; v8f v; };
  FAcc acc[2][2];
#pragma unroll
  for (int i = 0; i < 2; ++i)
#pragma unroll
    for (int j = 0; j < 2; ++j)
#pragma unroll
      for (int v = 0; v < 8; ++v) acc[i][j].f[v] = 0.0f;

  union Frag { U4 u[2]; v16bf v; };

  const int NK = K >> 5;             // K is always a multiple of 32 here

  // prologue: stage 0 -> buffer 0
  ASYNC_CP_B128(ldsA,       aptr);
  ASYNC_CP_B128(ldsA + 16u, aptr + 8);
  ASYNC_CP_B128(ldsB,       bptr);

  for (int k = 0; k < NK; ++k) {
    // stage-k data landed in LDS; make it visible to all waves
    asm volatile("s_wait_asynccnt 0" ::: "memory");
    __syncthreads();

    // overlap: kick off stage k+1 into the other buffer (safe: everyone has
    // finished reading it -- stage k-1 compute completed before this barrier)
    if (k + 1 < NK) {
      const bf16_t* an = aptr + ((k + 1) << 5);
      const bf16_t* bn = bptr + ((k + 1) << 5);
      const uint32_t abuf = ldsA + (uint32_t)((k + 1) & 1) * strA;
      const uint32_t bbuf = ldsB + (uint32_t)((k + 1) & 1) * strB;
      ASYNC_CP_B128(abuf,       an);
      ASYNC_CP_B128(abuf + 16u, an + 8);
      ASYNC_CP_B128(bbuf,       bn);
      if (k + 2 < NK) {                      // warm L2 for the stage after
        __builtin_prefetch(aptr + ((k + 2) << 5), 0, 2);
        __builtin_prefetch(bptr + ((k + 2) << 5), 0, 2);
      }
    }

    const int cb = k & 1;
    Frag fa[2], fb[2];
#pragma unroll
    for (int t = 0; t < 2; ++t) {
      int rA = wm + t * 16 + lr;
      fa[t].u[0] = *(const U4*)&lA[cb][rA][kh];
      fa[t].u[1] = *(const U4*)&lA[cb][rA][kh + 16];
      int rB = wn + t * 16 + lr;
      fb[t].u[0] = *(const U4*)&lB[cb][rB][kh];
      fb[t].u[1] = *(const U4*)&lB[cb][rB][kh + 16];
    }
#pragma unroll
    for (int i = 0; i < 2; ++i)
#pragma unroll
      for (int j = 0; j < 2; ++j)
        acc[i][j].v = __builtin_amdgcn_wmma_f32_16x16x32_bf16(
            false, fa[i].v, false, fb[j].v, (short)0, acc[i][j].v, false, false);
  }

  // epilogue: C layout VGPR v -> row v (+8 for lanes>=16), col = lane&15
  const int rowAdd = (lane >> 4) << 3;
  const int colL   = lane & 15;
  const int bb     = bias2 ? (batch % bias2_mod) : 0;
#pragma unroll
  for (int i = 0; i < 2; ++i)
#pragma unroll
    for (int j = 0; j < 2; ++j) {
      int gcol = nbase + wn + j * 16 + colL;
      if (gcol >= N) continue;
      float badd = 0.0f;
      if (bias)  badd += bias[(size_t)batch * (size_t)sBiasB + gcol];
      if (bias2) badd += bias2[(size_t)bb * bias2_ld + gcol];
      for (int v = 0; v < 8; ++v) {
        int grow = mbase + wm + i * 16 + rowAdd + v;
        if (grow >= M) continue;
        float r = acc[i][j].f[v] + badd;
        if (act == 1) r = fmaxf(r, 0.0f);
        size_t ci = (size_t)batch * (size_t)sCb + (size_t)grow * ldc + gcol;
        if (Cb) Cb[ci] = f2bf(r); else Cf[ci] = r;
      }
    }
}

// ---------------- squash over the 6 capsules ----------------
// capsule 0 = h2[task 5] (current), capsules 1..5 = h2[tasks 0..4]
__global__ void k_squash(const float* __restrict__ h2, bf16_t* __restrict__ aspPad,
                         bf16_t* __restrict__ feats) {
  int i = blockIdx.x * blockDim.x + threadIdx.x;
  if (i >= BB * SS * HH) return;
  const size_t MH = (size_t)MROWS * HH;
  float c[T1];
  c[0] = h2[(size_t)TT * MH + i];
#pragma unroll
  for (int t = 0; t < TT; ++t) c[1 + t] = h2[(size_t)t * MH + i];
  float sq = 1e-16f;
#pragma unroll
  for (int t = 0; t < T1; ++t) sq += c[t] * c[t];
  float scale = (sq / (1.0f + sq)) * rsqrtf(sq);
  int b   = i / (SS * HH);
  int rem = i - b * SS * HH;
  int s   = rem / HH;
  int h   = rem - s * HH;
  aspPad[((size_t)b * (SS + 6) + (s + 3)) * HH + h] = f2bf(c[0] * scale);
#pragma unroll
  for (int t = 0; t < TT; ++t)
    feats[(((size_t)t * BB + b) * SS + s) * (size_t)HH + h] = f2bf(c[1 + t] * scale);
}

// ---------------- aspect: relu+maxpool over length ----------------
__global__ void k_aspect_reduce(const float* __restrict__ o3, const float* __restrict__ o4,
                                const float* __restrict__ o5, float* __restrict__ caps2) {
  int i = blockIdx.x * blockDim.x + threadIdx.x;
  if (i >= 3 * BB * CO) return;
  int kidx = i / (BB * CO);
  int r    = i - kidx * BB * CO;
  int b    = r / CO, co = r - (r / CO) * CO;
  const float* src = (kidx == 0) ? o3 : ((kidx == 1) ? o4 : o5);
  int L = SS + kidx;                       // 128,129,130
  float m = -3.4e38f;
  for (int p = 0; p < L; ++p) m = fmaxf(m, src[((size_t)b * L + p) * CO + co]);
  m = fmaxf(m, 0.0f);
  caps2[((size_t)b * T1) * 300 + kidx * CO + co] = m;
}

// asp = aspect_feat @ fca_w^T + fca_b
__global__ void k_asp_proj(const float* __restrict__ caps2, const float* __restrict__ fcaw,
                           const float* __restrict__ fcab, float* __restrict__ aspProj) {
  int i = blockIdx.x * blockDim.x + threadIdx.x;
  if (i >= BB * CO) return;
  int b = i / CO, co = i - b * CO;
  float acc = fcab[co];
  const float* cf = caps2 + (size_t)b * T1 * 300;
  const float* wr = fcaw + (size_t)co * 300;
  for (int j = 0; j < 300; ++j) acc += cf[j] * wr[j];
  aspProj[i] = acc;
}

// z = max_p tanh(g1) * relu(g2)   (g2 already has bias + asp from GEMM epilogue)
__global__ void k_gated_reduce(const float* __restrict__ g1, const float* __restrict__ g2,
                               int L, int colOff, float* __restrict__ caps2) {
  int i = blockIdx.x * blockDim.x + threadIdx.x;
  if (i >= TT * BB * CO) return;
  int t = i / (BB * CO);
  int r = i - t * (BB * CO);
  int b = r / CO, co = r - (r / CO) * CO;
  size_t base = ((size_t)(t * BB + b)) * (size_t)L * CO + co;
  float m = -3.4e38f;
  for (int p = 0; p < L; ++p) {
    float zk = tanhf(g1[base + (size_t)p * CO]);
    float yk = fmaxf(g2[base + (size_t)p * CO], 0.0f);
    m = fmaxf(m, zk * yk);
  }
  caps2[((size_t)b * T1 + 1 + t) * 300 + colOff + co] = m;
}

// priors[n][b][i][s] = sum_c caps2[b][i][c] * route_w[n][i][c][s]
__global__ void k_priors(const float* __restrict__ caps2, const float* __restrict__ routew,
                         float* __restrict__ priors) {
  int i = blockIdx.x * blockDim.x + threadIdx.x;
  if (i >= NN * BB * T1 * SC) return;
  int s = i % SC; int r = i / SC;
  int cap = r % T1; r /= T1;
  int b = r % BB; int n = r / BB;
  const float* cv = caps2 + ((size_t)b * T1 + cap) * 300;
  const float* rw = routew + ((size_t)(n * T1 + cap)) * 300 * SC + s;
  float acc = 0.0f;
  for (int c = 0; c < 300; ++c) acc += cv[c] * rw[(size_t)c * SC];
  priors[i] = acc;
}

// 3-iteration dynamic routing, fully in registers; stores final (pre-squash) vote
__global__ void k_routing(const float* __restrict__ priors, float* __restrict__ vote) {
  int i = blockIdx.x * blockDim.x + threadIdx.x;
  if (i >= NN * BB) return;
  float pr[T1][SC];
  const float* p0 = priors + (size_t)i * T1 * SC;
#pragma unroll
  for (int c = 0; c < T1; ++c)
#pragma unroll
    for (int s = 0; s < SC; ++s) pr[c][s] = p0[c * SC + s];
  float logit[T1] = {0, 0, 0, 0, 0, 0};
  float v[SC];
  for (int it = 0; it < 3; ++it) {
    float mx = logit[0];
#pragma unroll
    for (int c = 1; c < T1; ++c) mx = fmaxf(mx, logit[c]);
    float e[T1], sum = 0.0f;
#pragma unroll
    for (int c = 0; c < T1; ++c) { e[c] = __expf(logit[c] - mx); sum += e[c]; }
    float inv = 1.0f / sum;
#pragma unroll
    for (int s = 0; s < SC; ++s) {
      float a = 0.0f;
#pragma unroll
      for (int c = 0; c < T1; ++c) a += e[c] * pr[c][s];
      v[s] = a * inv;
    }
    if (it < 2) {
      float sq = 1e-16f;
#pragma unroll
      for (int s = 0; s < SC; ++s) sq += v[s] * v[s];
      float sc = (sq / (1.0f + sq)) * rsqrtf(sq);
#pragma unroll
      for (int c = 0; c < T1; ++c) {
        float d = 0.0f;
#pragma unroll
        for (int s = 0; s < SC; ++s) d += pr[c][s] * v[s] * sc;
        logit[c] += d;
      }
    }
  }
  float* vo = vote + (size_t)i * SC;
#pragma unroll
  for (int s = 0; s < SC; ++s) vo[s] = v[s];
}

// out = x + h_cur + vote.reshape(B,1,H) broadcast over S
__global__ void k_final(const float* __restrict__ x, const float* __restrict__ h2,
                        const float* __restrict__ vote, float* __restrict__ out) {
  int i = blockIdx.x * blockDim.x + threadIdx.x;
  if (i >= BB * SS * HH) return;
  int b = i / (SS * HH);
  int h = i % HH;
  out[i] = x[i] + h2[(size_t)TT * MROWS * HH + i] + vote[(size_t)b * HH + h];
}

// ---------------- host ----------------
static inline int cdiv(int a, int b) { return (a + b - 1) / b; }

extern "C" void kernel_launch(void* const* d_in, const int* in_sizes, int n_in,
                              void* d_out, int out_size, void* d_ws, size_t ws_size,
                              hipStream_t stream) {
  const float* x    = (const float*)d_in[0];
  const float* fc1w = (const float*)d_in[1];
  const float* fc1b = (const float*)d_in[2];
  const float* fc2w = (const float*)d_in[3];
  const float* fc2b = (const float*)d_in[4];
  const float* c1w[3] = {(const float*)d_in[5],  (const float*)d_in[7],  (const float*)d_in[9]};
  const float* c1b[3] = {(const float*)d_in[6],  (const float*)d_in[8],  (const float*)d_in[10]};
  const float* c2w[3] = {(const float*)d_in[11], (const float*)d_in[13], (const float*)d_in[15]};
  const float* c2b[3] = {(const float*)d_in[12], (const float*)d_in[14], (const float*)d_in[16]};
  const float* c3w[3] = {(const float*)d_in[17], (const float*)d_in[19], (const float*)d_in[21]};
  const float* c3b[3] = {(const float*)d_in[18], (const float*)d_in[20], (const float*)d_in[22]};
  const float* fcaw   = (const float*)d_in[23];
  const float* fcab   = (const float*)d_in[24];
  const float* routew = (const float*)d_in[25];

  char* wsb = (char*)d_ws;
  size_t off = 0;
  auto alloc = [&](size_t bytes) -> void* {
    off = (off + 255) & ~(size_t)255;
    void* p = wsb + off;
    off += bytes;
    return p;
  };

  bf16_t* xbf    = (bf16_t*)alloc((size_t)MROWS * HH * 2);
  bf16_t* w1bf   = (bf16_t*)alloc((size_t)T1 * AA * HH * 2);
  bf16_t* w2bf   = (bf16_t*)alloc((size_t)T1 * HH * AA * 2);
  bf16_t* h1bf   = (bf16_t*)alloc((size_t)T1 * MROWS * AA * 2);
  float*  h2f    = (float*) alloc((size_t)T1 * MROWS * HH * 4);
  bf16_t* aspPad = (bf16_t*)alloc((size_t)BB * (SS + 6) * HH * 2);
  bf16_t* feats  = (bf16_t*)alloc((size_t)TT * BB * SS * HH * 2);
  bf16_t *wt1[3], *wt2[3], *wt3[3];
  float  *aspOut[3], *g1buf[3], *g2buf[3];
  for (int i = 0; i < 3; ++i) {
    int Kw = 3 + i;
    wt1[i] = (bf16_t*)alloc((size_t)CO * Kw * HH * 2);
    wt2[i] = (bf16_t*)alloc((size_t)CO * Kw * HH * 2);
    wt3[i] = (bf16_t*)alloc((size_t)CO * Kw * HH * 2);
  }
  for (int i = 0; i < 3; ++i) aspOut[i] = (float*)alloc((size_t)BB * (SS + i) * CO * 4);
  float* caps2   = (float*)alloc((size_t)BB * T1 * 300 * 4);
  float* aspProj = (float*)alloc((size_t)BB * CO * 4);
  for (int i = 0; i < 3; ++i) {
    int L = SS - (3 + i) + 1;   // 126,125,124
    g1buf[i] = (float*)alloc((size_t)TT * BB * L * CO * 4);
    g2buf[i] = (float*)alloc((size_t)TT * BB * L * CO * 4);
  }
  float* priors = (float*)alloc((size_t)NN * BB * T1 * SC * 4);
  float* vote   = (float*)alloc((size_t)NN * BB * SC * 4);

  // zero the time-padded aspect buffer (pad rows must be 0)
  hipMemsetAsync(aspPad, 0, (size_t)BB * (SS + 6) * HH * 2, stream);

  // prep conversions
  int n = MROWS * HH;
  k_f32_to_bf16<<<cdiv(n, BD), BD, 0, stream>>>(x, xbf, n);
  n = T1 * AA * HH;
  k_f32_to_bf16<<<cdiv(n, BD), BD, 0, stream>>>(fc1w, w1bf, n);
  k_f32_to_bf16<<<cdiv(n, BD), BD, 0, stream>>>(fc2w, w2bf, n);
  for (int i = 0; i < 3; ++i) {
    int Kw = 3 + i, m = CO * HH * Kw;
    k_convw<<<cdiv(m, BD), BD, 0, stream>>>(c1w[i], wt1[i], Kw);
    k_convw<<<cdiv(m, BD), BD, 0, stream>>>(c2w[i], wt2[i], Kw);
    k_convw<<<cdiv(m, BD), BD, 0, stream>>>(c3w[i], wt3[i], Kw);
  }

  // adapters: h1 = relu(x W1^T + b1), h2 = relu(h1 W2^T + b2), batched over 6 tasks
  {
    dim3 g(AA / 64, MROWS / 128, T1);
    k_wmma_gemm<<<g, BD, 0, stream>>>(xbf, 0, HH, w1bf, (long long)AA * HH, HH,
                                      nullptr, h1bf, (long long)MROWS * AA, AA,
                                      fc1b, AA, nullptr, 1, CO,
                                      MROWS, AA, HH, 1);
  }
  {
    dim3 g(HH / 64, MROWS / 128, T1);
    k_wmma_gemm<<<g, BD, 0, stream>>>(h1bf, (long long)MROWS * AA, AA, w2bf,
                                      (long long)HH * AA, AA,
                                      h2f, nullptr, (long long)MROWS * HH, HH,
                                      fc2b, HH, nullptr, 1, CO,
                                      MROWS, HH, AA, 1);
  }

  // squash 6 capsules -> padded aspect capsule (bf16) + prev-task feats (bf16)
  n = BB * SS * HH;
  k_squash<<<cdiv(n, BD), BD, 0, stream>>>(h2f, aspPad, feats);

  // aspect convs (implicit GEMM, pad K-2 via buffer offset)
  for (int i = 0; i < 3; ++i) {
    int Kw = 3 + i, L = SS + Kw - 3;
    dim3 g(2, cdiv(L, 128), BB);
    k_wmma_gemm<<<g, BD, 0, stream>>>(aspPad + (size_t)(5 - Kw) * HH,
                                      (long long)(SS + 6) * HH, HH,
                                      wt3[i], 0, Kw * HH,
                                      aspOut[i], nullptr, (long long)L * CO, CO,
                                      c3b[i], 0, nullptr, 1, CO,
                                      L, CO, Kw * HH, 0);
  }
  n = 3 * BB * CO;
  k_aspect_reduce<<<cdiv(n, BD), BD, 0, stream>>>(aspOut[0], aspOut[1], aspOut[2], caps2);
  n = BB * CO;
  k_asp_proj<<<cdiv(n, BD), BD, 0, stream>>>(caps2, fcaw, fcab, aspProj);

  // gated convs over the 5 previous-task capsules (batch = 320)
  for (int i = 0; i < 3; ++i) {
    int Kw = 3 + i, L = SS - Kw + 1;
    dim3 g(2, 1, TT * BB);
    k_wmma_gemm<<<g, BD, 0, stream>>>(feats, (long long)SS * HH, HH,
                                      wt1[i], 0, Kw * HH,
                                      g1buf[i], nullptr, (long long)L * CO, CO,
                                      c1b[i], 0, nullptr, 1, CO,
                                      L, CO, Kw * HH, 0);
    k_wmma_gemm<<<g, BD, 0, stream>>>(feats, (long long)SS * HH, HH,
                                      wt2[i], 0, Kw * HH,
                                      g2buf[i], nullptr, (long long)L * CO, CO,
                                      c2b[i], 0, aspProj, BB, CO,
                                      L, CO, Kw * HH, 0);
  }
  n = TT * BB * CO;
  for (int i = 0; i < 3; ++i)
    k_gated_reduce<<<cdiv(n, BD), BD, 0, stream>>>(g1buf[i], g2buf[i], SS - (3 + i) + 1,
                                                   100 * i, caps2);

  // routing
  n = NN * BB * T1 * SC;
  k_priors<<<cdiv(n, BD), BD, 0, stream>>>(caps2, routew, priors);
  n = NN * BB;
  k_routing<<<cdiv(n, BD), BD, 0, stream>>>(priors, vote);

  // final residual add
  n = BB * SS * HH;
  k_final<<<cdiv(n, BD), BD, 0, stream>>>(x, h2f, vote, (float*)d_out);
}
